// Decoder_901943132836
// MI455X (gfx1250) — matrix operand order repaired and checked by reading, not compiled, not run on hardware
//
#include <hip/hip_runtime.h>
#include <hip/hip_bf16.h>
#include <math.h>

typedef _Float16 v16h __attribute__((ext_vector_type(16)));
typedef _Float16 h8   __attribute__((ext_vector_type(8)));
typedef float    v8f  __attribute__((ext_vector_type(8)));

namespace cfg {
constexpr int T_ = 512, NB = 128, L_ = 256;
constexpr int KV = 128, VS = 128, EMB = 256, HID = 512, VOC = 34, VOCP = 48;
constexpr int XW  = EMB + VS + HID;   // 896  : [emb | ctx | h1]
constexpr int G1W = 4 * HID;          // 2048
constexpr int HW  = HID + KV;         // 640  : [h1 | h2]
constexpr int G2W = 4 * KV;           // 512
constexpr int OCW = KV + VS;          // 256  : [h2 | ctx]
constexpr int NBLK = 16, TPB = 256, NTHR = NBLK * TPB;   // 128 waves

constexpr size_t a256(size_t x) { return (x + 255) & ~size_t(255); }
constexpr size_t OFF_BAR = 0;                                        // 2 uints (256B slot)
constexpr size_t OFF_W1T = 256;                                      // f16 (896 x 2048)
constexpr size_t OFF_W2T = a256(OFF_W1T + size_t(XW) * G1W * 2);     // f16 (640 x 512)
constexpr size_t OFF_WOT = a256(OFF_W2T + size_t(HW) * G2W * 2);     // f16 (256 x 48)
constexpr size_t OFF_B1  = a256(OFF_WOT + size_t(OCW) * VOCP * 2);   // f32 2048
constexpr size_t OFF_B2  = a256(OFF_B1 + size_t(G1W) * 4);           // f32 512
constexpr size_t OFF_BO  = a256(OFF_B2 + size_t(G2W) * 4);           // f32 48
constexpr size_t OFF_H1  = a256(OFF_BO + size_t(VOCP) * 4);          // f32 128x512
constexpr size_t OFF_C1  = OFF_H1 + size_t(NB) * HID * 4;
constexpr size_t OFF_H2  = OFF_C1 + size_t(NB) * HID * 4;            // f32 128x128
constexpr size_t OFF_C2  = OFF_H2 + size_t(NB) * KV * 4;
constexpr size_t OFF_G1  = OFF_C2 + size_t(NB) * KV * 4;             // f32 128x2048
constexpr size_t OFF_G2  = OFF_G1 + size_t(NB) * G1W * 4;            // f32 128x512
constexpr size_t OFF_EN  = OFF_G2 + size_t(NB) * G2W * 4;            // f32 128x512
constexpr size_t OFF_AT  = OFF_EN + size_t(NB) * T_ * 4;             // f32 128x512
constexpr size_t OFF_XC  = OFF_AT + size_t(NB) * T_ * 4;             // f16 128x896
constexpr size_t OFF_HC  = OFF_XC + size_t(NB) * XW * 2;             // f16 128x640
constexpr size_t OFF_OC  = OFF_HC + size_t(NB) * HW * 2;             // f16 128x256
constexpr size_t OFF_CTX = OFF_OC + size_t(NB) * OCW * 2;            // f16 128x128
} // namespace cfg

using namespace cfg;

__device__ __forceinline__ float sigf(float x) { return 1.0f / (1.0f + __expf(-x)); }

// Load a 16-half WMMA fragment from two contiguous 16B chunks.
__device__ __forceinline__ v16h ldfrag(const _Float16* p0, const _Float16* p1) {
  h8 x = *(const h8*)p0;
  h8 y = *(const h8*)p1;
  return __builtin_shufflevector(x, y, 0,1,2,3,4,5,6,7,8,9,10,11,12,13,14,15);
}

__device__ __forceinline__ v8f wmma16(v16h a, v16h b, v8f c) {
  return __builtin_amdgcn_wmma_f32_16x16x32_f16(false, a, false, b, (short)0, c, false, false);
}

// Device-scope sense(generation) grid barrier; bar[0]=arrive cnt, bar[1]=release gen.
__device__ __forceinline__ void gbar(unsigned* bar, unsigned* genS) {
  __threadfence();            // drain this thread's stores to device scope
  __syncthreads();
  if (threadIdx.x == 0) {
    unsigned g = ++(*genS);
    unsigned prev = __hip_atomic_fetch_add(&bar[0], 1u, __ATOMIC_ACQ_REL, __HIP_MEMORY_SCOPE_AGENT);
    if (prev == (unsigned)NBLK - 1u) {
      __hip_atomic_store(&bar[0], 0u, __ATOMIC_RELAXED, __HIP_MEMORY_SCOPE_AGENT);
      __hip_atomic_store(&bar[1], g, __ATOMIC_RELEASE, __HIP_MEMORY_SCOPE_AGENT);
    } else {
      while (__hip_atomic_load(&bar[1], __ATOMIC_ACQUIRE, __HIP_MEMORY_SCOPE_AGENT) != g)
        __builtin_amdgcn_s_sleep(2);
    }
  }
  __syncthreads();
  __threadfence();            // acquire: invalidate near caches before reading peers' data
}

// ---------- prep: zero barrier/state, fuse biases, transpose weights to f16 B-layout ----------
__global__ void prep_kernel(const float* __restrict__ Wih1, const float* __restrict__ Whh1,
                            const float* __restrict__ bih1, const float* __restrict__ bhh1,
                            const float* __restrict__ Wih2, const float* __restrict__ Whh2,
                            const float* __restrict__ bih2, const float* __restrict__ bhh2,
                            const float* __restrict__ Wout, const float* __restrict__ bout,
                            char* __restrict__ ws) {
  _Float16* W1t = (_Float16*)(ws + OFF_W1T);
  _Float16* W2t = (_Float16*)(ws + OFF_W2T);
  _Float16* Wot = (_Float16*)(ws + OFF_WOT);
  float* b1 = (float*)(ws + OFF_B1);
  float* b2 = (float*)(ws + OFF_B2);
  float* bo = (float*)(ws + OFF_BO);
  float* h1 = (float*)(ws + OFF_H1);
  float* c1 = (float*)(ws + OFF_C1);
  float* h2 = (float*)(ws + OFF_H2);
  float* c2 = (float*)(ws + OFF_C2);
  _Float16* ctxh = (_Float16*)(ws + OFF_CTX);
  unsigned* bar = (unsigned*)(ws + OFF_BAR);

  const size_t R_W1 = size_t(XW) * G1W;           // 1,835,008
  const size_t R_W2 = size_t(HW) * G2W;           // 327,680
  const size_t R_WO = size_t(OCW) * VOCP;         // 12,288
  const size_t R_B1 = G1W, R_B2 = G2W, R_BO = VOCP;
  const size_t R_H  = size_t(NB) * HID;           // h1 & c1
  const size_t R_K  = size_t(NB) * KV;            // h2, c2, ctx
  const size_t TOTAL = R_W1 + R_W2 + R_WO + R_B1 + R_B2 + R_BO + 2 * R_H + 3 * R_K + 64;

  for (size_t i = size_t(blockIdx.x) * blockDim.x + threadIdx.x; i < TOTAL;
       i += size_t(gridDim.x) * blockDim.x) {
    size_t j = i;
    if (j < R_W1) {                 // W1t[k][n] = [Wih1 | Whh1]^T
      size_t k = j / G1W, n = j - k * G1W;
      float v = (k < size_t(EMB + VS)) ? Wih1[n * (EMB + VS) + k] : Whh1[n * HID + (k - (EMB + VS))];
      W1t[j] = (_Float16)v; continue;
    }
    j -= R_W1;
    if (j < R_W2) {                 // W2t[k][n] = [Wih2 | Whh2]^T
      size_t k = j / G2W, n = j - k * G2W;
      float v = (k < size_t(HID)) ? Wih2[n * HID + k] : Whh2[n * KV + (k - HID)];
      W2t[j] = (_Float16)v; continue;
    }
    j -= R_W2;
    if (j < R_WO) {                 // Wot[k][n] = Wout^T padded to 48 cols
      size_t k = j / VOCP, n = j - k * VOCP;
      W1t = W1t; // no-op
      Wot[j] = (_Float16)((n < size_t(VOC)) ? Wout[n * OCW + k] : 0.0f); continue;
    }
    j -= R_WO;
    if (j < R_B1) { b1[j] = bih1[j] + bhh1[j]; continue; }
    j -= R_B1;
    if (j < R_B2) { b2[j] = bih2[j] + bhh2[j]; continue; }
    j -= R_B2;
    if (j < R_BO) { bo[j] = (j < size_t(VOC)) ? bout[j] : 0.0f; continue; }
    j -= R_BO;
    if (j < R_H) { h1[j] = 0.0f; continue; }
    j -= R_H;
    if (j < R_H) { c1[j] = 0.0f; continue; }
    j -= R_H;
    if (j < R_K) { h2[j] = 0.0f; continue; }
    j -= R_K;
    if (j < R_K) { c2[j] = 0.0f; continue; }
    j -= R_K;
    if (j < R_K) { ctxh[j] = (_Float16)0.0f; continue; }
    j -= R_K;
    bar[j] = 0u;                    // 64 uints incl. arrive/release
  }
}

// ---------- persistent decoder: full L=256 scan in one launch ----------
__global__ void __launch_bounds__(TPB, 1)
decoder_kernel(const float* __restrict__ key, const float* __restrict__ values,
               const int* __restrict__ lens, const int* __restrict__ text,
               const float* __restrict__ emb, char* __restrict__ ws,
               float* __restrict__ out) {
  __shared__ unsigned genS;
  if (threadIdx.x == 0) genS = 0u;
  __syncthreads();

  unsigned* bar = (unsigned*)(ws + OFF_BAR);
  const _Float16* W1t = (const _Float16*)(ws + OFF_W1T);
  const _Float16* W2t = (const _Float16*)(ws + OFF_W2T);
  const _Float16* Wot = (const _Float16*)(ws + OFF_WOT);
  const float* b1 = (const float*)(ws + OFF_B1);
  const float* b2 = (const float*)(ws + OFF_B2);
  const float* bo = (const float*)(ws + OFF_BO);
  float* h1 = (float*)(ws + OFF_H1);
  float* c1 = (float*)(ws + OFF_C1);
  float* h2 = (float*)(ws + OFF_H2);
  float* c2 = (float*)(ws + OFF_C2);
  float* g1 = (float*)(ws + OFF_G1);
  float* g2 = (float*)(ws + OFF_G2);
  float* en = (float*)(ws + OFF_EN);
  float* at = (float*)(ws + OFF_AT);
  _Float16* Xc   = (_Float16*)(ws + OFF_XC);
  _Float16* Hc   = (_Float16*)(ws + OFF_HC);
  _Float16* OC   = (_Float16*)(ws + OFF_OC);
  _Float16* ctxh = (_Float16*)(ws + OFF_CTX);

  const int tid  = blockIdx.x * TPB + threadIdx.x;  // 0..4095
  const int lane = threadIdx.x & 31;
  const int w    = tid >> 5;                        // global wave 0..127
  const int half = lane >> 4;
  const int lm   = lane & 15;

  for (int t = 0; t < L_; ++t) {
    // ---- Phase A: Xcat = [emb[text[:,t]] | ctx | h1]  (f16) ----
    for (int p = 0; p < (NB * XW) / NTHR; ++p) {    // 28
      int idx = tid + p * NTHR;
      int n = idx / XW, k = idx - n * XW;
      float v;
      if (k < EMB)            v = emb[size_t(text[n * L_ + t]) * EMB + k];
      else if (k < EMB + VS)  v = (float)ctxh[n * VS + (k - EMB)];
      else                    v = h1[size_t(n) * HID + (k - EMB - VS)];
      Xc[idx] = (_Float16)v;
    }
    gbar(bar, &genS);

    // ---- Phase B: g1 = Xcat @ W1t  (WMMA f32.16x16x32.f16) ----
    {
      const int n0 = w * 16;                        // 128 N-tiles
      v8f acc[8] = {};
      for (int kt = 0; kt < XW / 32; ++kt) {        // 28
        const _Float16* brow = W1t + size_t(kt * 32 + half * 8) * G1W + n0 + lm;
        v16h bf;
        #pragma unroll
        for (int s = 0; s < 16; ++s) bf[s] = brow[size_t(s + (s >> 3) * 8) * G1W];
        #pragma unroll
        for (int m = 0; m < 8; ++m) {
          const _Float16* arow = Xc + size_t(m * 16 + lm) * XW + kt * 32 + half * 8;
          v16h af = ldfrag(arow, arow + 16);
          acc[m] = wmma16(af, bf, acc[m]);
        }
      }
      #pragma unroll
      for (int m = 0; m < 8; ++m)
        #pragma unroll
        for (int r = 0; r < 8; ++r)
          g1[size_t(m * 16 + half * 8 + r) * G1W + n0 + lm] = acc[m][r];
    }
    gbar(bar, &genS);

    // ---- Phase C: LSTM1 gates -> h1,c1 ; Hcat = [h1 | h2_prev] ----
    for (int p = 0; p < (NB * HID) / NTHR; ++p) {   // 16
      int idx = tid * 16 + p;
      int n = idx >> 9, u = idx & 511;
      size_t rb = size_t(n) * G1W;
      float gi = g1[rb + u]           + b1[u];
      float gf = g1[rb + HID + u]     + b1[HID + u];
      float gg = g1[rb + 2 * HID + u] + b1[2 * HID + u];
      float go = g1[rb + 3 * HID + u] + b1[3 * HID + u];
      float c = sigf(gf) * c1[size_t(n) * HID + u] + sigf(gi) * tanhf(gg);
      float h = sigf(go) * tanhf(c);
      c1[size_t(n) * HID + u] = c;
      h1[size_t(n) * HID + u] = h;
      Hc[size_t(n) * HW + u] = (_Float16)h;
    }
    for (int p = 0; p < (NB * KV) / NTHR; ++p) {    // 4 : copy prev h2 into Hcat
      int idx = tid * 4 + p;
      int n = idx >> 7, v = idx & 127;
      Hc[size_t(n) * HW + HID + v] = (_Float16)h2[size_t(n) * KV + v];
    }
    gbar(bar, &genS);

    // ---- Phase D: g2 = Hcat @ W2t (WMMA) ----
    {
      const int nt = w & 31, n0 = nt * 16;
      const int mb = (w >> 5) * 2;
      v8f acc[2] = {};
      for (int kt = 0; kt < HW / 32; ++kt) {        // 20
        const _Float16* brow = W2t + size_t(kt * 32 + half * 8) * G2W + n0 + lm;
        v16h bf;
        #pragma unroll
        for (int s = 0; s < 16; ++s) bf[s] = brow[size_t(s + (s >> 3) * 8) * G2W];
        #pragma unroll
        for (int mm = 0; mm < 2; ++mm) {
          const _Float16* arow = Hc + size_t((mb + mm) * 16 + lm) * HW + kt * 32 + half * 8;
          v16h af = ldfrag(arow, arow + 16);
          acc[mm] = wmma16(af, bf, acc[mm]);
        }
      }
      #pragma unroll
      for (int mm = 0; mm < 2; ++mm)
        #pragma unroll
        for (int r = 0; r < 8; ++r)
          g2[size_t((mb + mm) * 16 + half * 8 + r) * G2W + n0 + lm] = acc[mm][r];
    }
    gbar(bar, &genS);

    // ---- Phase E: LSTM2 gates -> h2,c2 ; OC[:, :128] = f16(h2) ----
    for (int p = 0; p < (NB * KV) / NTHR; ++p) {    // 4
      int idx = tid * 4 + p;
      int n = idx >> 7, v = idx & 127;
      size_t rb = size_t(n) * G2W;
      float gi = g2[rb + v]          + b2[v];
      float gf = g2[rb + KV + v]     + b2[KV + v];
      float gg = g2[rb + 2 * KV + v] + b2[2 * KV + v];
      float go = g2[rb + 3 * KV + v] + b2[3 * KV + v];
      float c = sigf(gf) * c2[size_t(n) * KV + v] + sigf(gi) * tanhf(gg);
      float h = sigf(go) * tanhf(c);
      c2[size_t(n) * KV + v] = c;
      h2[size_t(n) * KV + v] = h;
      OC[size_t(n) * OCW + v] = (_Float16)h;
    }
    gbar(bar, &genS);

    // ---- Phase F: energy[n][tt] = key[tt,n,:] . h2[n,:]  (masked) ----
    for (int p = 0; p < (NB * T_) / NTHR; ++p) {    // 16
      int idx = tid * 16 + p;
      int n = idx >> 9, tt = idx & 511;
      const float4* kp = (const float4*)(key + (size_t(tt) * NB + n) * KV);
      const float4* hp = (const float4*)(h2 + size_t(n) * KV);
      float s = 0.0f;
      #pragma unroll 8
      for (int q = 0; q < KV / 4; ++q) {
        float4 a = kp[q], b = hp[q];
        s += a.x * b.x + a.y * b.y + a.z * b.z + a.w * b.w;
      }
      en[size_t(n) * T_ + tt] = (tt >= lens[n]) ? -1e9f : s;
    }
    gbar(bar, &genS);

    // ---- Phase G: softmax over T, one wave per batch row ----
    {
      const int n = w;
      float vals[16];
      float mx = -3.4e38f;
      #pragma unroll
      for (int j = 0; j < 16; ++j) {
        vals[j] = en[size_t(n) * T_ + lane + j * 32];
        mx = fmaxf(mx, vals[j]);
      }
      for (int o = 16; o >= 1; o >>= 1) mx = fmaxf(mx, __shfl_xor(mx, o, 32));
      float s = 0.0f;
      #pragma unroll
      for (int j = 0; j < 16; ++j) { vals[j] = __expf(vals[j] - mx); s += vals[j]; }
      for (int o = 16; o >= 1; o >>= 1) s += __shfl_xor(s, o, 32);
      float inv = 1.0f / s;
      #pragma unroll
      for (int j = 0; j < 16; ++j) at[size_t(n) * T_ + lane + j * 32] = vals[j] * inv;
    }
    gbar(bar, &genS);

    // ---- Phase H: ctx[n][v] = sum_t attn * values[t,n,v] ; OC[:,128:] ----
    {
      const int n = tid >> 5;
      const int v0 = (tid & 31) * 4;
      float ax = 0.f, ay = 0.f, az = 0.f, aw = 0.f;
      const float* arow = at + size_t(n) * T_;
      for (int tt = 0; tt < T_; ++tt) {
        float a = arow[tt];
        float4 vv = *(const float4*)(values + (size_t(tt) * NB + n) * VS + v0);
        ax += a * vv.x; ay += a * vv.y; az += a * vv.z; aw += a * vv.w;
      }
      ctxh[size_t(n) * VS + v0 + 0] = (_Float16)ax;
      ctxh[size_t(n) * VS + v0 + 1] = (_Float16)ay;
      ctxh[size_t(n) * VS + v0 + 2] = (_Float16)az;
      ctxh[size_t(n) * VS + v0 + 3] = (_Float16)aw;
      OC[size_t(n) * OCW + KV + v0 + 0] = (_Float16)ax;
      OC[size_t(n) * OCW + KV + v0 + 1] = (_Float16)ay;
      OC[size_t(n) * OCW + KV + v0 + 2] = (_Float16)az;
      OC[size_t(n) * OCW + KV + v0 + 3] = (_Float16)aw;
    }
    gbar(bar, &genS);

    // ---- Phase I: pred = [h2|ctx] @ Wout^T + b_out  (WMMA, VOC padded 48) ----
    if (w < 24) {                                   // 8 M-tiles x 3 N-tiles
      const int mt = w / 3, nt = w - mt * 3, n0 = nt * 16;
      v8f acc = {};
      for (int kt = 0; kt < OCW / 32; ++kt) {       // 8
        const _Float16* brow = Wot + size_t(kt * 32 + half * 8) * VOCP + n0 + lm;
        v16h bf;
        #pragma unroll
        for (int s = 0; s < 16; ++s) bf[s] = brow[size_t(s + (s >> 3) * 8) * VOCP];
        const _Float16* arow = OC + size_t(mt * 16 + lm) * OCW + kt * 32 + half * 8;
        v16h af = ldfrag(arow, arow + 16);
        acc = wmma16(af, bf, acc);
      }
      const int vcol = n0 + lm;
      if (vcol < VOC) {
        float bb = bo[vcol];
        #pragma unroll
        for (int r = 0; r < 8; ++r) {
          int nb = mt * 16 + half * 8 + r;
          out[(size_t(nb) * L_ + t) * VOC + vcol] = acc[r] + bb;
        }
      }
    }
    // next-step Phase A only reads data already synced (h1, ctx); the sync
    // after Phase A orders Phase I of step t before Phase B of step t+1.
  }
}

extern "C" void kernel_launch(void* const* d_in, const int* in_sizes, int n_in,
                              void* d_out, int out_size, void* d_ws, size_t ws_size,
                              hipStream_t stream) {
  (void)in_sizes; (void)n_in; (void)out_size; (void)ws_size;
  const float* key    = (const float*)d_in[0];
  const float* values = (const float*)d_in[1];
  const int*   lens   = (const int*)d_in[2];
  const int*   text   = (const int*)d_in[3];
  const float* emb    = (const float*)d_in[4];
  const float* Wih1   = (const float*)d_in[5];
  const float* Whh1   = (const float*)d_in[6];
  const float* bih1   = (const float*)d_in[7];
  const float* bhh1   = (const float*)d_in[8];
  const float* Wih2   = (const float*)d_in[9];
  const float* Whh2   = (const float*)d_in[10];
  const float* bih2   = (const float*)d_in[11];
  const float* bhh2   = (const float*)d_in[12];
  const float* Wout   = (const float*)d_in[13];
  const float* bout   = (const float*)d_in[14];
  char* ws = (char*)d_ws;

  prep_kernel<<<1024, 256, 0, stream>>>(Wih1, Whh1, bih1, bhh1,
                                        Wih2, Whh2, bih2, bhh2,
                                        Wout, bout, ws);
  decoder_kernel<<<NBLK, TPB, 0, stream>>>(key, values, lens, text, emb, ws,
                                           (float*)d_out);
}
